// GAT_24094766531339
// MI455X (gfx1250) — compile-verified, matrix operand order checked by reading, not run on hardware
//
#include <hip/hip_runtime.h>
#include <hip/hip_bf16.h>

// ---------------------------------------------------------------------------
// 2-layer GAT (DGL GATConv semantics) + mean-pool + linear + softmax
// MI455X / gfx1250:
//   * GEMMs via v_wmma_f32_16x16x32_bf16, operands pre-packed into the ISA's
//     wave32 lane layouts so the inner loop is b128 loads + WMMA only.
//   * Edge softmax via L2-resident f32 atomics (working set ~120MB < 192MB L2).
// ---------------------------------------------------------------------------

typedef __attribute__((ext_vector_type(16))) __bf16 v16bf;
typedef __attribute__((ext_vector_type(8)))  __bf16 v8bf;
typedef __attribute__((ext_vector_type(8)))  float  v8f;

#define NEG_SLOPE  0.2f
#define SELU_ALPHA 1.6732632423543772f
#define SELU_SCALE 1.0507009873554805f

__device__ __forceinline__ float leakyf(float x) { return x > 0.f ? x : NEG_SLOPE * x; }
__device__ __forceinline__ float seluf(float x) {
  return x > 0.f ? SELU_SCALE * x : SELU_SCALE * SELU_ALPHA * (__expf(x) - 1.f);
}
// order-preserving float -> u32 key (monotone): works with atomicMax(u32)
__device__ __forceinline__ unsigned fkey(float f) {
  int b = __float_as_int(f);
  return (unsigned)(b ^ ((b >> 31) | 0x80000000));
}
__device__ __forceinline__ float fdekey(unsigned k) {
  int b = (k & 0x80000000u) ? (int)(k ^ 0x80000000u) : ~(int)k;
  return __int_as_float(b);
}

// --------------------------- f32 -> bf16 (row-major) -----------------------
__global__ void cvt_f32_bf16(const float* __restrict__ s, __bf16* __restrict__ d, int n) {
  int i = blockIdx.x * blockDim.x + threadIdx.x;
  if (i < n) d[i] = (__bf16)s[i];
}

// ---------------- pack B[K x Nc] f32 -> WMMA-native bf16 layout ------------
// Bpack[kt][nt][lane][j] = B[(kt*32 + kj) * Nc + nt*16 + (lane&15)]
// kj = koff + j + (j>=8 ? 8 : 0), koff = (lane>=16)*8  (16-bit B 32x16 layout)
__global__ void pack_b_wmma(const float* __restrict__ B, __bf16* __restrict__ Bp,
                            int K, int Nc) {
  int i = blockIdx.x * blockDim.x + threadIdx.x;
  if (i >= K * Nc) return;
  const int j    = i & 15;
  const int lane = (i >> 4) & 31;
  const int t    = i >> 9;
  const int tilesN = Nc >> 4;
  const int nt   = t % tilesN;
  const int kt   = t / tilesN;
  const int koff = (lane >> 4) << 3;
  const int kj   = koff + j + ((j >= 8) ? 8 : 0);
  Bp[i] = (__bf16)B[(size_t)(kt * 32 + kj) * Nc + nt * 16 + (lane & 15)];
}

// --------------------------- WMMA bf16 GEMM --------------------------------
// C[M x Nc] = A[M x K] (bf16 row-major) @ B (pre-packed bf16), f32 accumulate.
// One wave32 computes a 16x32 C tile (A fragment reused by 2 WMMAs).
// M mult of 16, Nc mult of 32, K mult of 32. Uniform early-out only, so EXEC
// is all-ones around every WMMA as the ISA requires.
__global__ void gemm_wmma_bf16(const __bf16* __restrict__ A, const __bf16* __restrict__ Bp,
                               float* __restrict__ C, int M, int K, int Nc) {
  const int lane    = threadIdx.x & 31;
  const int wave    = blockIdx.x * (blockDim.x >> 5) + (threadIdx.x >> 5);
  const int tilesN  = Nc >> 4;
  const int tilesN2 = Nc >> 5;
  const int tiles   = (M >> 4) * tilesN2;
  if (wave >= tiles) return;
  const int tm   = wave / tilesN2;
  const int tn2  = wave % tilesN2;          // pair of 16-col tiles
  const int l15  = lane & 15;
  const int koff = (lane >> 4) << 3;        // 0 (lanes 0-15) or 8 (lanes 16-31)
  const __bf16* arow = A + (size_t)(tm * 16 + l15) * K;

  v8f acc0 = {}, acc1 = {};
  for (int kt = 0; kt < (K >> 5); ++kt) {
    const int kb = kt << 5;
    // A 16x32 bf16 layout: two contiguous 8-elem runs per lane -> 2x b128
    v8bf alo = *(const v8bf*)(arow + kb + koff);
    v8bf ahi = *(const v8bf*)(arow + kb + koff + 16);
    v16bf a = __builtin_shufflevector(alo, ahi, 0, 1, 2, 3, 4, 5, 6, 7,
                                      8, 9, 10, 11, 12, 13, 14, 15);
    // packed B: each lane's 16 elems are contiguous (32B)
    const __bf16* bp = Bp + (((size_t)kt * tilesN + tn2 * 2) * 32 + lane) * 16;
    v16bf b0 = *(const v16bf*)bp;
    v16bf b1 = *(const v16bf*)(bp + 32 * 16);
    acc0 = __builtin_amdgcn_wmma_f32_16x16x32_bf16(false, a, false, b0,
                                                   (short)0, acc0, false, false);
    acc1 = __builtin_amdgcn_wmma_f32_16x16x32_bf16(false, a, false, b1,
                                                   (short)0, acc1, false, false);
  }
  // f32 C 16x16: VGPR r -> M = r (lanes 0-15) / 8+r (lanes 16-31)
  float* c0 = C + (size_t)(tm * 16 + koff) * Nc + tn2 * 32 + l15;
#pragma unroll
  for (int r = 0; r < 8; ++r) {
    c0[(size_t)r * Nc]      = acc0[r];
    c0[(size_t)r * Nc + 16] = acc1[r];
  }
}

// --------------------- per-(node,channel) attention dots -------------------
// fs: [N, NH, 32]; attn indexed by head = c & 3 (works for NH=4 and NH=16).
__global__ void attn_dots(const float* __restrict__ fs, const float* __restrict__ al,
                          const float* __restrict__ ar, float* __restrict__ el,
                          float* __restrict__ er, int N, int NH) {
  int i = blockIdx.x * blockDim.x + threadIdx.x;
  if (i >= N * NH) return;
  const int c = i % NH;
  const float* f = fs + (size_t)i * 32;
  const float* a = al + (c & 3) * 32;
  const float* b = ar + (c & 3) * 32;
  float sl = 0.f, sr = 0.f;
#pragma unroll
  for (int d = 0; d < 32; ++d) { float v = f[d]; sl += v * a[d]; sr += v * b[d]; }
  el[i] = sl;
  er[i] = sr;
}

// --------------------------- edge softmax passes ---------------------------
__global__ void edge_max(const int* __restrict__ src, const int* __restrict__ dst,
                         const float* __restrict__ el, const float* __restrict__ er,
                         unsigned* __restrict__ mkey, int E, int NH) {
  int i = blockIdx.x * blockDim.x + threadIdx.x;
  if (i >= E * NH) return;
  const int e = i / NH, c = i % NH;
  const int s = src[e], d = dst[e];
  const float v = leakyf(el[s * NH + c] + er[d * NH + c]);
  atomicMax(&mkey[d * NH + c], fkey(v));
}

__global__ void edge_denom(const int* __restrict__ src, const int* __restrict__ dst,
                           const float* __restrict__ el, const float* __restrict__ er,
                           const unsigned* __restrict__ mkey, float* __restrict__ denom,
                           int E, int NH) {
  int i = blockIdx.x * blockDim.x + threadIdx.x;
  if (i >= E * NH) return;
  const int e = i / NH, c = i % NH;
  const int s = src[e], d = dst[e];
  const unsigned k = mkey[d * NH + c];
  const float m  = k ? fdekey(k) : 0.f;     // key 0 == -inf sentinel (zero-deg)
  const float ee = __expf(leakyf(el[s * NH + c] + er[d * NH + c]) - m);
  atomicAdd(&denom[d * NH + c], ee);
}

__global__ void edge_aggr(const int* __restrict__ src, const int* __restrict__ dst,
                          const float* __restrict__ el, const float* __restrict__ er,
                          const unsigned* __restrict__ mkey, const float* __restrict__ denom,
                          const float* __restrict__ fs, float* __restrict__ rst,
                          int E, int NH) {
  int i = blockIdx.x * blockDim.x + threadIdx.x;
  if (i >= E * NH) return;
  const int e = i / NH, c = i % NH;
  const int s = src[e], d = dst[e];
  const unsigned k = mkey[d * NH + c];
  const float m     = k ? fdekey(k) : 0.f;
  const float ee    = __expf(leakyf(el[s * NH + c] + er[d * NH + c]) - m);
  const float alpha = ee / fmaxf(denom[d * NH + c], 1e-9f);
  const float4* fp = (const float4*)(fs + ((size_t)s * NH + c) * 32);
  float* rp = rst + ((size_t)d * NH + c) * 32;
#pragma unroll
  for (int q = 0; q < 8; ++q) {
    float4 v = fp[q];
    atomicAdd(&rp[q * 4 + 0], alpha * v.x);
    atomicAdd(&rp[q * 4 + 1], alpha * v.y);
    atomicAdd(&rp[q * 4 + 2], alpha * v.z);
    atomicAdd(&rp[q * 4 + 3], alpha * v.w);
  }
}

// bias (broadcast over [.., H, 32], head = c&3) + SELU; optional bf16 copy
// (feeds the next layer's WMMA GEMM directly). In-place safe for f32 path.
__global__ void bias_selu(const float* __restrict__ rst, const float* __restrict__ bias,
                          float* __restrict__ h, __bf16* __restrict__ hb, int N, int NH) {
  int i = blockIdx.x * blockDim.x + threadIdx.x;
  if (i >= N * NH * 32) return;
  const int d = i & 31;
  const int c = (i >> 5) % NH;
  const float v = seluf(rst[i] + bias[(c & 3) * 32 + d]);
  h[i] = v;
  if (hb) hb[i] = (__bf16)v;
}

// mean over the 16 (prefix x head) channels -> [N, 32]
__global__ void mean_pool(const float* __restrict__ h2, float* __restrict__ pooled, int N) {
  int i = blockIdx.x * blockDim.x + threadIdx.x;
  if (i >= N * 32) return;
  const int n = i >> 5, d = i & 31;
  float s = 0.f;
#pragma unroll
  for (int c = 0; c < 16; ++c) s += h2[((size_t)n * 16 + c) * 32 + d];
  pooled[i] = s * (1.f / 16.f);
}

// logits = pooled @ Wc + bc ; row softmax. One thread per node (40x32 dots).
__global__ void classify(const float* __restrict__ pooled, const float* __restrict__ Wc,
                         const float* __restrict__ bc, float* __restrict__ out, int N) {
  int n = blockIdx.x * blockDim.x + threadIdx.x;
  if (n >= N) return;
  float p[32];
#pragma unroll
  for (int d = 0; d < 32; ++d) p[d] = pooled[(size_t)n * 32 + d];
  float lg[40];
  float mx = -1e30f;
  for (int c = 0; c < 40; ++c) {
    float s = bc[c];
#pragma unroll
    for (int d = 0; d < 32; ++d) s += p[d] * Wc[d * 40 + c];
    lg[c] = s;
    mx = fmaxf(mx, s);
  }
  float sum = 0.f;
  for (int c = 0; c < 40; ++c) { float e = __expf(lg[c] - mx); lg[c] = e; sum += e; }
  const float inv = 1.f / sum;
  for (int c = 0; c < 40; ++c) out[(size_t)n * 40 + c] = lg[c] * inv;
}

// ---------------------------------------------------------------------------
static inline int dcv(long n, int b) { return (int)((n + b - 1) / b); }

extern "C" void kernel_launch(void* const* d_in, const int* in_sizes, int n_in,
                              void* d_out, int out_size, void* d_ws, size_t ws_size,
                              hipStream_t stream) {
  const float* x    = (const float*)d_in[0];
  const int*   src0 = (const int*)d_in[1];
  const int*   dst0 = (const int*)d_in[2];
  const int*   src1 = (const int*)d_in[3];
  const int*   dst1 = (const int*)d_in[4];
  const float* W1   = (const float*)d_in[5];
  const float* al1  = (const float*)d_in[6];
  const float* ar1  = (const float*)d_in[7];
  const float* b1   = (const float*)d_in[8];
  const float* W2   = (const float*)d_in[9];
  const float* al2  = (const float*)d_in[10];
  const float* ar2  = (const float*)d_in[11];
  const float* b2   = (const float*)d_in[12];
  const float* Wc   = (const float*)d_in[13];
  const float* bc   = (const float*)d_in[14];

  const int N = in_sizes[0] / 128;   // 20000 (multiple of 16)
  const int E = in_sizes[1];         // 160000

  // ---- workspace carve-up (~122 MB total; L2-resident on MI455X) ----
  char* p = (char*)d_ws;
  auto carve = [&](size_t bytes) -> char* {
    char* r = p;
    p += (bytes + 255) & ~(size_t)255;
    return r;
  };
  __bf16*   W1p  = (__bf16*)  carve((size_t)128 * 128 * 2);  // packed WMMA layout
  __bf16*   W2p  = (__bf16*)  carve((size_t)32 * 128 * 2);
  __bf16*   xb   = (__bf16*)  carve((size_t)N * 128 * 2);    // x as bf16
  __bf16*   h1b  = (__bf16*)  carve((size_t)N * 128 * 2);    // selu(h1) as bf16
  float*    fs1  = (float*)   carve((size_t)N * 128 * 4);    // [N,4,32]
  float*    el1  = (float*)   carve((size_t)N * 4 * 4);
  float*    er1  = (float*)   carve((size_t)N * 4 * 4);
  unsigned* m1   = (unsigned*)carve((size_t)N * 4 * 4);
  float*    den1 = (float*)   carve((size_t)N * 4 * 4);
  float*    h1   = (float*)   carve((size_t)N * 128 * 4);    // rst1 -> selu in-place
  float*    fs2  = (float*)   carve((size_t)N * 512 * 4);    // [N,4,4,32]
  float*    el2  = (float*)   carve((size_t)N * 16 * 4);
  float*    er2  = (float*)   carve((size_t)N * 16 * 4);
  unsigned* m2   = (unsigned*)carve((size_t)N * 16 * 4);
  float*    den2 = (float*)   carve((size_t)N * 16 * 4);
  float*    h2   = (float*)   carve((size_t)N * 512 * 4);    // rst2 -> selu in-place
  float*    pool = (float*)   carve((size_t)N * 32 * 4);
  (void)ws_size;

  const int B = 256;

  // operand prep: pack weights into WMMA lane layout, x -> bf16
  pack_b_wmma<<<dcv(128 * 128, B), B, 0, stream>>>(W1, W1p, 128, 128);
  pack_b_wmma<<<dcv(32 * 128, B), B, 0, stream>>>(W2, W2p, 32, 128);
  cvt_f32_bf16<<<dcv((long)N * 128, B), B, 0, stream>>>(x, xb, N * 128);

  // ---------------- layer 1 ----------------
  {
    const int tiles = (N / 16) * (128 / 32);              // 5000 waves (625 blocks)
    gemm_wmma_bf16<<<dcv(tiles, 8), B, 0, stream>>>(xb, W1p, fs1, N, 128, 128);
  }
  attn_dots<<<dcv((long)N * 4, B), B, 0, stream>>>(fs1, al1, ar1, el1, er1, N, 4);
  hipMemsetAsync(m1, 0, (size_t)N * 4 * 4, stream);
  hipMemsetAsync(den1, 0, (size_t)N * 4 * 4, stream);
  hipMemsetAsync(h1, 0, (size_t)N * 128 * 4, stream);
  edge_max<<<dcv((long)E * 4, B), B, 0, stream>>>(src0, dst0, el1, er1, m1, E, 4);
  edge_denom<<<dcv((long)E * 4, B), B, 0, stream>>>(src0, dst0, el1, er1, m1, den1, E, 4);
  edge_aggr<<<dcv((long)E * 4, B), B, 0, stream>>>(src0, dst0, el1, er1, m1, den1, fs1, h1, E, 4);
  bias_selu<<<dcv((long)N * 128, B), B, 0, stream>>>(h1, b1, h1, h1b, N, 4);

  // ---------------- layer 2 ----------------
  {
    const int tiles = ((N * 4) / 16) * (128 / 32);        // 20000 waves (2500 blocks)
    gemm_wmma_bf16<<<dcv(tiles, 8), B, 0, stream>>>(h1b, W2p, fs2, N * 4, 32, 128);
  }
  attn_dots<<<dcv((long)N * 16, B), B, 0, stream>>>(fs2, al2, ar2, el2, er2, N, 16);
  hipMemsetAsync(m2, 0, (size_t)N * 16 * 4, stream);
  hipMemsetAsync(den2, 0, (size_t)N * 16 * 4, stream);
  hipMemsetAsync(h2, 0, (size_t)N * 512 * 4, stream);
  edge_max<<<dcv((long)E * 16, B), B, 0, stream>>>(src1, dst1, el2, er2, m2, E, 16);
  edge_denom<<<dcv((long)E * 16, B), B, 0, stream>>>(src1, dst1, el2, er2, m2, den2, E, 16);
  edge_aggr<<<dcv((long)E * 16, B), B, 0, stream>>>(src1, dst1, el2, er2, m2, den2, fs2, h2, E, 16);
  bias_selu<<<dcv((long)N * 512, B), B, 0, stream>>>(h2, b2, h2, (__bf16*)nullptr, N, 16);

  // ---------------- head ----------------
  mean_pool<<<dcv((long)N * 32, B), B, 0, stream>>>(h2, pool, N);
  classify<<<dcv(N, B), B, 0, stream>>>(pool, Wc, bc, (float*)d_out, N);
}
